// MultiHeadAttention_70497593196785
// MI455X (gfx1250) — compile-verified, matrix operand order checked by reading, not run on hardware
//
#include <hip/hip_runtime.h>
#include <hip/hip_bf16.h>
#include <math.h>

// ---------------------------------------------------------------------------
// Causal MHA forward for MI455X (gfx1250, wave32, WMMA).
//   B=4, L=2048, D=1024, H=16, DH=64.  q=k=v = x.view(B,L,H,DH).
//   d_out = [ out (B*L*H*DH floats, flat (B,L,H,DH) order) | attn (B*H*L*L) ]
// Memory-bound: attn is a 1 GiB streaming write (~46us @ 23.3 TB/s); the two
// bf16-WMMA GEMMs (~6.9e10 FLOPs, f32 accumulate) cost far less.  x (32 MB)
// stays resident in the 192 MB L2, so per-lane gathers are cheap; attn stores
// use nontemporal hints so the streaming output does not evict x from L2.
// ---------------------------------------------------------------------------

#define B_   4
#define L_   2048
#define D_   1024
#define H_   16
#define DH_  64
#define QROWS 16          // query rows per workgroup
#define NWAVES 8          // 256 threads = 8 wave32
#define SCALE 0.125f      // 1/sqrt(64)
#define MASKV (-1e9f)

typedef __attribute__((ext_vector_type(16))) __bf16 v16bf;
typedef __attribute__((ext_vector_type(8)))  float  v8f;
typedef __attribute__((ext_vector_type(4)))  float  f32x4;

// WMMA bf16 16x16x32 A/B per-lane element layout (ISA 7.12.2, 16-bit A 16x32):
//   lane = 16*half + idx16 ; element j (0..15) maps to
//   K(j) = 16*(j>>3) + 8*half + (j&7)          (M or N = idx16)
// => per lane the 16 elements are two runs of 8 contiguous K values.
// Load each run as two float4 (32B, aligned) and convert to bf16.
__device__ __forceinline__ v16bf frag_rowmajor(const float* __restrict__ rowp,
                                               int half, float s) {
  const float* p0 = rowp + 8 * half;        // j = 0..7
  const float* p1 = rowp + 16 + 8 * half;   // j = 8..15
  f32x4 a0 = *(const f32x4*)(p0);
  f32x4 a1 = *(const f32x4*)(p0 + 4);
  f32x4 a2 = *(const f32x4*)(p1);
  f32x4 a3 = *(const f32x4*)(p1 + 4);
  v16bf r;
#pragma unroll
  for (int i = 0; i < 4; ++i) {
    r[i]      = (__bf16)(a0[i] * s);
    r[4 + i]  = (__bf16)(a1[i] * s);
    r[8 + i]  = (__bf16)(a2[i] * s);
    r[12 + i] = (__bf16)(a3[i] * s);
  }
  return r;
}

__global__ void mha_causal_wmma_kernel(const float* __restrict__ x,
                                       float* __restrict__ outp,
                                       float* __restrict__ attnp) {
  extern __shared__ float smem[];
  float* S  = smem;                 // 16 x 2048 fp32 scores/probs  (128 KB)
  float* OB = smem + QROWS * L_;    // 2 x 16 x 64 fp32 partial out (8 KB)

  const int qt = blockIdx.x & (L_ / QROWS - 1);   // query tile 0..127
  const int bh = blockIdx.x >> 7;                 // b*H + h
  const int b  = bh >> 4;
  const int h  = bh & (H_ - 1);
  const int q0 = qt * QROWS;

  const int tid = threadIdx.x;
  const int w   = tid >> 5;    // wave 0..7
  const int ln  = tid & 31;    // lane 0..31
  const int lh  = ln >> 4;     // half 0/1
  const int l16 = ln & 15;     // M (A/C) or N (B/C)

  // zero the partial-out buffers
  for (int i = tid; i < 2 * QROWS * DH_; i += NWAVES * 32) OB[i] = 0.0f;

  // qkv[b,h,l,d] = x[(b*L + l)*D + h*DH + d]
  const float* xb = x + (size_t)b * L_ * D_ + h * DH_;

  // ---------------- Phase 1: scores = (Q*scale) @ K^T, causal tiles only ---
  // A fragments for this block's 16 Q rows; two k-steps cover DH=64.
  const float* qrow = xb + (size_t)(q0 + l16) * D_;
  v16bf aq0 = frag_rowmajor(qrow,      lh, SCALE);
  v16bf aq1 = frag_rowmajor(qrow + 32, lh, SCALE);

  for (int nt = w; nt <= qt; nt += NWAVES) {
    // B operand: B[k][n] = Kmat[n][k] -> lane n reads Kmat row nt*16+n
    const float* krow = xb + (size_t)(nt * QROWS + l16) * D_;
    v16bf bk0 = frag_rowmajor(krow,      lh, 1.0f);
    v16bf bk1 = frag_rowmajor(krow + 32, lh, 1.0f);
    v8f c = {};
    c = __builtin_amdgcn_wmma_f32_16x16x32_bf16(false, aq0, false, bk0,
                                                (short)0, c, false, false);
    c = __builtin_amdgcn_wmma_f32_16x16x32_bf16(false, aq1, false, bk1,
                                                (short)0, c, false, false);
    // C/D layout: element v in lane -> M = v + 8*half, N = l16
    const int col = nt * QROWS + l16;
#pragma unroll
    for (int v = 0; v < 8; ++v) {
      const int m = v + 8 * lh;
      S[m * L_ + col] = (col <= q0 + m) ? c[v] : MASKV;
    }
  }
  __syncthreads();

  // ---------------- Phase 2: row softmax (fp32), write attn ----------------
  // Vectorized float4 LDS traffic; attn written with nontemporal b128 stores
  // (streaming 1 GiB output must not evict L2-resident x).
  const int blockKend  = (qt + 1) * QROWS;           // cols touched by WMMA
  const int nChunks    = (blockKend + 31) >> 5;      // 32-wide k-chunks
  const int blockKend2 = nChunks << 5;               // padded to chunk size
  float* attnBH = attnp + ((size_t)bh * L_ + q0) * L_;

#pragma unroll
  for (int rr = 0; rr < 2; ++rr) {
    const int r     = 2 * w + rr;                    // wave w owns rows 2w,2w+1
    const int ncols = q0 + r + 1;                    // causal row length
    float* Srow = S + r * L_;
    float* arow = attnBH + (size_t)r * L_;

    float mx = -INFINITY;
    for (int c0 = 4 * ln; c0 < ncols; c0 += 128) {
      f32x4 v = *(const f32x4*)(Srow + c0);          // LDS alloc covers row
#pragma unroll
      for (int j = 0; j < 4; ++j)
        mx = fmaxf(mx, (c0 + j < ncols) ? v[j] : -INFINITY);
    }
#pragma unroll
    for (int off = 16; off; off >>= 1) mx = fmaxf(mx, __shfl_xor(mx, off, 32));

    float sum = 0.0f;
    for (int c0 = 4 * ln; c0 < ncols; c0 += 128) {
      f32x4 v = *(const f32x4*)(Srow + c0);
#pragma unroll
      for (int j = 0; j < 4; ++j)
        sum += (c0 + j < ncols) ? __expf(v[j] - mx) : 0.0f;
    }
#pragma unroll
    for (int off = 16; off; off >>= 1) sum += __shfl_xor(sum, off, 32);
    const float inv = 1.0f / sum;

    // normalize: write P back to LDS (incl. zeroed chunk tail) + attn out
    for (int c0 = 4 * ln; c0 < blockKend2; c0 += 128) {
      f32x4 v = *(const f32x4*)(Srow + c0);
      f32x4 p;
#pragma unroll
      for (int j = 0; j < 4; ++j)
        p[j] = (c0 + j < ncols) ? __expf(v[j] - mx) * inv : 0.0f;
      *(f32x4*)(Srow + c0) = p;
      __builtin_nontemporal_store(p, (f32x4*)(arow + c0));
    }
    const f32x4 z = {};
    for (int c0 = blockKend2 + 4 * ln; c0 < L_; c0 += 128)
      __builtin_nontemporal_store(z, (f32x4*)(arow + c0));
  }
  __syncthreads();

  // ---------------- Phase 3: out = P @ V -----------------------------------
  // 4 column tiles (DH=64) x 2 k-halves = 8 waves, no atomics.
  const int ntile = w & 3;
  const int kh    = w >> 2;
  const int colOff = ntile * QROWS + l16;            // d-column this lane owns
  v8f acc = {};
  for (int c = kh; c < nChunks; c += 2) {
    // A: P rows (fp32 in LDS, contiguous k) -> bf16 fragment
    v16bf a = frag_rowmajor(S + l16 * L_ + c * 32, lh, 1.0f);
    // B: B[k][n] = V[c*32 + K(j)][colOff]; strided rows, L2-resident
    v16bf bb;
#pragma unroll
    for (int j = 0; j < 16; ++j) {
      const int k   = ((j >> 3) << 4) + (lh << 3) + (j & 7);
      const int row = c * 32 + k;
      bb[j] = (__bf16)xb[(size_t)row * D_ + colOff];
    }
    acc = __builtin_amdgcn_wmma_f32_16x16x32_bf16(false, a, false, bb,
                                                  (short)0, acc, false, false);
  }
#pragma unroll
  for (int v = 0; v < 8; ++v) {
    const int m = v + 8 * lh;
    OB[kh * (QROWS * DH_) + m * DH_ + ntile * QROWS + l16] = acc[v];
  }
  __syncthreads();

  // combine the two k-halves; out flat layout is (B, L, H, DH)
  for (int i = tid; i < QROWS * DH_; i += NWAVES * 32) {
    const int r = i >> 6;
    const int d = i & (DH_ - 1);
    outp[(((size_t)b * L_ + q0 + r) * H_ + h) * DH_ + d] =
        OB[i] + OB[QROWS * DH_ + i];
  }
}

extern "C" void kernel_launch(void* const* d_in, const int* in_sizes, int n_in,
                              void* d_out, int out_size, void* d_ws, size_t ws_size,
                              hipStream_t stream) {
  (void)in_sizes; (void)n_in; (void)d_ws; (void)ws_size; (void)out_size;
  const float* x = (const float*)d_in[0];   // W_q/W_k/W_v (d_in[1..3]) are dead
                                            // code; mask (d_in[4]) is causal ->
                                            // computed analytically.
  float* outp  = (float*)d_out;
  float* attnp = outp + (size_t)B_ * L_ * H_ * DH_;

  const dim3 grid(B_ * H_ * (L_ / QROWS));  // 8192 blocks
  const dim3 block(NWAVES * 32);            // 8 wave32
  const size_t shmem = (QROWS * L_ + 2 * QROWS * DH_) * sizeof(float); // 136 KB

  mha_causal_wmma_kernel<<<grid, block, shmem, stream>>>(x, outp, attnp);
}
